// GraphAttentionLayer_34986803593494
// MI455X (gfx1250) — matrix operand order repaired and checked by reading, compiled, not run on hardware
//
#include <hip/hip_runtime.h>

// ---------------------------------------------------------------------------
// Types for CDNA5 WMMA (wave32): 16x16x32 bf16 -> f32
// ---------------------------------------------------------------------------
typedef __attribute__((ext_vector_type(16))) __bf16        v16bf;
typedef __attribute__((ext_vector_type(8)))  float         v8f;
typedef __attribute__((ext_vector_type(4)))  float         f4;
typedef __attribute__((ext_vector_type(4)))  int           i4;
typedef __attribute__((ext_vector_type(4)))  unsigned int  u4;

union BfPack { unsigned int u[8]; v16bf v; };

#if __has_builtin(__builtin_amdgcn_cvt_pk_bf16_f32)
__device__ __forceinline__ unsigned int pack2bf(float lo, float hi) {
  auto r = __builtin_amdgcn_cvt_pk_bf16_f32(lo, hi);   // v_cvt_pk_bf16_f32
  union { decltype(r) v; unsigned int u; } c;
  c.v = r;
  return c.u;
}
#else
__device__ __forceinline__ unsigned int pack2bf(float lo, float hi) {
  // round-to-nearest-even fp32 -> bf16, packed pair (software fallback)
  unsigned int a = __float_as_uint(lo), b = __float_as_uint(hi);
  a = (a + 0x7FFFu + ((a >> 16) & 1u)) >> 16;
  b = (b + 0x7FFFu + ((b >> 16) & 1u)) >> 16;
  return (a & 0xFFFFu) | (b << 16);
}
#endif

#define N_ROWS   8192
#define FIN      256
#define FOUT     128
#define L2E      1.442695040888963f
#define NEG2     (-1.2986e16f)   /* NEG * log2(e) */

// ---------------------------------------------------------------------------
// Prep: fp32 -> bf16 bulk convert (8 elements / thread), off the hot path.
// ---------------------------------------------------------------------------
__global__ __launch_bounds__(256) void k_cvt(const float* __restrict__ src,
                                             unsigned short* __restrict__ dst,
                                             int n8) {
  long t = (long)blockIdx.x * 256 + threadIdx.x;
  if (t >= n8) return;
  const f4* s = (const f4*)(src + t * 8);
  f4 a = s[0], b = s[1];
  u4 o;
  o[0] = pack2bf(a[0], a[1]); o[1] = pack2bf(a[2], a[3]);
  o[2] = pack2bf(b[0], b[1]); o[3] = pack2bf(b[2], b[3]);
  *(u4*)(dst + t * 8) = o;
}

// ---------------------------------------------------------------------------
// Kernel 1: h = X @ W^T + b  (bf16 WMMA, f32 accum). Wave owns a 16x128 tile.
// A (16x32 bf16): lane holds row M=lane%16; K run 16h..16h+15 (= X row, contig).
// B (32x16 bf16): lane holds col N=lane%16; K run 16h..16h+15 (= W row, contig).
// All 8 B tiles are staged into registers BEFORE the WMMA chain so the loads
// pipeline instead of serializing on s_wait_loadcnt 0.
// Outputs: h (f32 row-major) and hbT (bf16 TRANSPOSED [f][j] for k_attn).
// ---------------------------------------------------------------------------
__global__ __launch_bounds__(128) void k_gemm_h(
    const unsigned short* __restrict__ Xb, const unsigned short* __restrict__ Wbf,
    const float* __restrict__ Wb, float* __restrict__ h,
    unsigned short* __restrict__ hbT) {
  const int lane = threadIdx.x & 31;
  const int wv   = threadIdx.x >> 5;
  const int row  = lane & 15;
  const int half = lane >> 4;
  const int i0   = blockIdx.x * 64 + wv * 16;

  v8f acc[8];
#pragma unroll
  for (int t = 0; t < 8; ++t) acc[t] = (v8f)0.0f;

  const int r1 = half * 16, r2 = half * 16 + 8;
  const unsigned short* xrow = Xb + (long)(i0 + row) * FIN;

  for (int k0 = 0; k0 < FIN; k0 += 32) {
    // ---- issue ALL loads for this K-step first ----
    u4 a0 = *(const u4*)(xrow + k0 + r1);   // A: 8 bf16, slots 0..3
    u4 a1 = *(const u4*)(xrow + k0 + r2);   // A: 8 bf16, slots 4..7
    BfPack Bt[8];
#pragma unroll
    for (int nt = 0; nt < 8; ++nt) {
      const unsigned short* wrow =
          Wbf + (long)(nt * 16 + row) * FIN + k0 + half * 16;
      u4 b0 = *(const u4*)(wrow);
      u4 b1 = *(const u4*)(wrow + 8);
#pragma unroll
      for (int i = 0; i < 4; ++i) { Bt[nt].u[i] = b0[i]; Bt[nt].u[4 + i] = b1[i]; }
    }
    BfPack A;
#pragma unroll
    for (int i = 0; i < 4; ++i) { A.u[i] = a0[i]; A.u[4 + i] = a1[i]; }

    // ---- WMMA chain ----
#pragma unroll
    for (int nt = 0; nt < 8; ++nt)
      acc[nt] = __builtin_amdgcn_wmma_f32_16x16x32_bf16(
          false, A.v, false, Bt[nt].v, (short)0, acc[nt], false, false);
  }

  // Epilogue: bias add; store h (f32 row-major) + hbT (bf16 transposed).
#pragma unroll
  for (int nt = 0; nt < 8; ++nt) {
    float bias = Wb[nt * 16 + row];
#pragma unroll
    for (int v = 0; v < 8; ++v) {
      float val = acc[nt][v] + bias;
      int   gm  = i0 + v + half * 8;        // global row index
      int   gn  = nt * 16 + row;            // global feature index
      h[(long)gm * FOUT + gn] = val;
      hbT[(long)gn * N_ROWS + gm] =
          (unsigned short)(pack2bf(val, 0.0f) & 0xFFFFu);
    }
  }
}

// ---------------------------------------------------------------------------
// Kernel 2: e1[i] = h[i,:].a1 ; e2p[j] = h[j,:].a2 + a_b.  Wave per row.
// ---------------------------------------------------------------------------
__global__ __launch_bounds__(256) void k_edges(
    const float* __restrict__ h, const float* __restrict__ aw,
    const float* __restrict__ ab, float* __restrict__ e1,
    float* __restrict__ e2p) {
  const int lane = threadIdx.x & 31;
  const int wid  = threadIdx.x >> 5;
  const long i   = (long)blockIdx.x * 8 + wid;
  f4 hv = *(const f4*)(h + i * FOUT + lane * 4);
  f4 a1 = *(const f4*)(aw + lane * 4);
  f4 a2 = *(const f4*)(aw + FOUT + lane * 4);
  float d1 = 0.f, d2 = 0.f;
#pragma unroll
  for (int t = 0; t < 4; ++t) { d1 += hv[t] * a1[t]; d2 += hv[t] * a2[t]; }
#pragma unroll
  for (int off = 16; off >= 1; off >>= 1) {
    d1 += __shfl_xor(d1, off, 32);
    d2 += __shfl_xor(d2, off, 32);
  }
  if (lane == 0) { e1[i] = d1; e2p[i] = d2 + ab[0]; }
}

// ---------------------------------------------------------------------------
// Kernel 3: fused masked-softmax attention @ h (flash-style, bf16 WMMA).
// Wave owns 16 rows x 128 output features. Per 32-col block:
//   1) issue ALL global loads (adj, e2p, 16x b128 B tiles) up front,
//   2) do the online-softmax VALU work (overlaps load latency),
//   3) run the 8-WMMA chain on pre-staged B registers.
// Online softmax in log2 domain; m init = NEG2 self-corrects masked prefixes
// and reproduces the reference's uniform softmax for all-masked rows.
// ---------------------------------------------------------------------------
__global__ __launch_bounds__(128) void k_attn(
    const int* __restrict__ adj, const unsigned short* __restrict__ hbT,
    const float* __restrict__ e1, const float* __restrict__ e2p,
    float* __restrict__ out) {
  const int lane = threadIdx.x & 31;
  const int wv   = threadIdx.x >> 5;
  const int row  = lane & 15;
  const int half = lane >> 4;
  const int i0   = blockIdx.x * 64 + wv * 16;
  const long gi  = i0 + row;

  float m = NEG2, lsum = 0.0f;
  v8f acc[8];
#pragma unroll
  for (int t = 0; t < 8; ++t) acc[t] = (v8f)0.0f;

  const float e1v  = e1[gi];
  const int   r1 = half * 16, r2 = half * 16 + 8;
  const int*  arow = adj + gi * (long)N_ROWS;

  for (int jb = 0; jb < N_ROWS; jb += 32) {
    // stream-prefetch the adjacency row ~32 iterations ahead
    __builtin_prefetch((const void*)(arow + jb + 1024), 0, 1);

    // ---- 1) issue all loads for this block up front ----
    i4 m0 = *(const i4*)(arow + jb + r1);
    i4 m1 = *(const i4*)(arow + jb + r1 + 4);
    i4 m2 = *(const i4*)(arow + jb + r2);
    i4 m3 = *(const i4*)(arow + jb + r2 + 4);
    f4 q0 = *(const f4*)(e2p + jb + r1);
    f4 q1 = *(const f4*)(e2p + jb + r1 + 4);
    f4 q2 = *(const f4*)(e2p + jb + r2);
    f4 q3 = *(const f4*)(e2p + jb + r2 + 4);

    BfPack Bt[8];
#pragma unroll
    for (int nt = 0; nt < 8; ++nt) {
      const unsigned short* bp =
          hbT + (long)(nt * 16 + row) * N_ROWS + jb + half * 16;
      u4 b0 = *(const u4*)(bp);
      u4 b1 = *(const u4*)(bp + 8);
#pragma unroll
      for (int i = 0; i < 4; ++i) { Bt[nt].u[i] = b0[i]; Bt[nt].u[4 + i] = b1[i]; }
    }

    // ---- 2) masked leaky-relu scores + online softmax (log2 domain) ----
    int am[16]; float ev[16];
#pragma unroll
    for (int i = 0; i < 4; ++i) {
      am[i] = m0[i]; am[4 + i] = m1[i]; am[8 + i] = m2[i]; am[12 + i] = m3[i];
      ev[i] = q0[i]; ev[4 + i] = q1[i]; ev[8 + i] = q2[i]; ev[12 + i] = q3[i];
    }

    float s[16], bm = NEG2;
#pragma unroll
    for (int t = 0; t < 16; ++t) {
      float x = e1v + ev[t];
      x = (x > 0.0f) ? x : 0.2f * x;       // leaky_relu(0.2)
      s[t] = (am[t] > 0) ? x * L2E : NEG2; // into log2 domain
      bm = fmaxf(bm, s[t]);
    }
    bm = fmaxf(bm, __shfl_xor(bm, 16, 32)); // row max across the two K-halves

    float mn    = fmaxf(m, bm);
    float scale = exp2f(m - mn);            // m==mn==NEG2 -> 1 (self-correcting)
    float ps    = 0.0f;
    BfPack A;
#pragma unroll
    for (int v = 0; v < 8; ++v) {
      float p0 = exp2f(s[2 * v] - mn);
      float p1 = exp2f(s[2 * v + 1] - mn);
      ps += p0 + p1;
      A.u[v] = pack2bf(p0, p1);
    }
    lsum = lsum * scale + ps;
    m = mn;

    // rescale acc: row M = v + 8*half lives in lane (v + 8*half)
    float rs[8];
#pragma unroll
    for (int v = 0; v < 8; ++v) rs[v] = __shfl(scale, v + half * 8, 32);
#pragma unroll
    for (int nt = 0; nt < 8; ++nt)
#pragma unroll
      for (int v = 0; v < 8; ++v) acc[nt][v] *= rs[v];

    // ---- 3) 8-WMMA chain on pre-staged B registers ----
#pragma unroll
    for (int nt = 0; nt < 8; ++nt)
      acc[nt] = __builtin_amdgcn_wmma_f32_16x16x32_bf16(
          false, A.v, false, Bt[nt].v, (short)0, acc[nt], false, false);
  }

  // normalize + ELU + store
  float tot = lsum + __shfl_xor(lsum, 16, 32);
  float rinv[8];
#pragma unroll
  for (int v = 0; v < 8; ++v) rinv[v] = 1.0f / __shfl(tot, v + half * 8, 32);
#pragma unroll
  for (int nt = 0; nt < 8; ++nt)
#pragma unroll
    for (int v = 0; v < 8; ++v) {
      float val = acc[nt][v] * rinv[v];
      val = (val > 0.0f) ? val : (__expf(val) - 1.0f);  // elu
      out[(long)(i0 + v + half * 8) * FOUT + nt * 16 + row] = val;
    }
}

// ---------------------------------------------------------------------------
extern "C" void kernel_launch(void* const* d_in, const int* in_sizes, int n_in,
                              void* d_out, int out_size, void* d_ws,
                              size_t ws_size, hipStream_t stream) {
  const float* X   = (const float*)d_in[0];  // 8192x256
  const int*   adj = (const int*)d_in[1];    // 8192x8192
  const float* Ww  = (const float*)d_in[2];  // 128x256
  const float* Wb  = (const float*)d_in[3];  // 128
  const float* aw  = (const float*)d_in[4];  // 1x256
  const float* ab  = (const float*)d_in[5];  // 1
  float* out = (float*)d_out;                // 8192x128

  char* ws = (char*)d_ws;
  unsigned short* Xb  = (unsigned short*)ws;                              // 4 MB
  float*          h   = (float*)(ws + (size_t)4 * 1024 * 1024);          // 4 MB
  unsigned short* hbT = (unsigned short*)(ws + (size_t)8 * 1024 * 1024); // 2 MB
  unsigned short* Wbf = (unsigned short*)(ws + (size_t)10 * 1024 * 1024);// 64 KB
  float*          e1  = (float*)(ws + (size_t)10 * 1024 * 1024 + 65536);
  float*          e2p = e1 + N_ROWS;

  // bf16 conversions (off the hot path)
  hipLaunchKernelGGL(k_cvt, dim3((N_ROWS * FIN / 8) / 256), dim3(256), 0,
                     stream, X, Xb, N_ROWS * FIN / 8);
  hipLaunchKernelGGL(k_cvt, dim3((FOUT * FIN / 8) / 256), dim3(256), 0,
                     stream, Ww, Wbf, FOUT * FIN / 8);

  hipLaunchKernelGGL(k_gemm_h, dim3(N_ROWS / 64), dim3(128), 0, stream,
                     Xb, Wbf, Wb, h, hbT);
  hipLaunchKernelGGL(k_edges, dim3(N_ROWS / 8), dim3(256), 0, stream,
                     h, aw, ab, e1, e2p);
  hipLaunchKernelGGL(k_attn, dim3(N_ROWS / 64), dim3(128), 0, stream,
                     adj, hbT, e1, e2p, out);
}
